// EdgeDecoder_15556371546274
// MI455X (gfx1250) — matrix operand order repaired and checked by reading, not compiled, not run on hardware
//
#include <hip/hip_runtime.h>

// ---------------------------------------------------------------------------
// EdgeDecoder for MI455X (gfx1250, wave32, WMMA)
//
// logits[e] = relu(concat(U[u[e]], V[v[e]]) @ W1 + b1) @ W2 + b2
// Two edge types (clicks, buys), E edges each. out = [clicks | buys].
//
// Strategy: gather fp32 activations -> bf16 A-fragments in LDS,
// pre-convert W1 -> bf16 B-fragments in d_ws (L2-resident),
// v_wmma_f32_16x16x32_bf16 with fp32 accumulation, fused ReLU+dot epilogue.
// ---------------------------------------------------------------------------

typedef __attribute__((ext_vector_type(16))) __bf16 v16bf;
typedef __attribute__((ext_vector_type(8)))  float  v8f;

union ABFrag {
    uint4 q[2];   // 32 bytes
    v16bf bf;
};

#define HID      256
#define IN_DIM   128
#define KDIM     256          // 2*IN_DIM
#define M_TILES  4            // 4 x 16 = 64 edges per workgroup
#define NT_PER_WAVE 4         // 16 n-tiles / 4 waves
#define W1_FRAG_DWORDS 32768  // 256*256 bf16 = 128KB = 32768 dwords per type

// round-to-nearest-even f32 -> bf16, packed pair -> u32 (lo = a, hi = b)
__device__ __forceinline__ unsigned packbf2(float a, float b) {
    unsigned ua = __float_as_uint(a);
    unsigned ub = __float_as_uint(b);
    ua += 0x7FFFu + ((ua >> 16) & 1u);
    ub += 0x7FFFu + ((ub >> 16) & 1u);
    return (ua >> 16) | (ub & 0xFFFF0000u);
}

// ---------------------------------------------------------------------------
// Prep: W1 (row-major [K=256][N=256] f32) -> bf16 B-fragment layout in ws.
// Fragment (kt, nt): lane L holds B[kt*32 + 16*(L/16) + i][nt*16 + L%16],
// i = 0..15, stored as 8 contiguous dwords per lane.
// ---------------------------------------------------------------------------
__global__ void edgemlp_prep_w1(const float* __restrict__ W1c,
                                const float* __restrict__ W1b,
                                unsigned* __restrict__ ws) {
    int g    = blockIdx.x * 256 + threadIdx.x;   // 0..8191
    int ty   = g >> 12;
    int r    = g & 4095;
    int kt   = r >> 9;        // 0..7
    int r2   = r & 511;
    int nt   = r2 >> 5;       // 0..15
    int lane = r2 & 31;
    int n    = nt * 16 + (lane & 15);
    int half = lane >> 4;

    const float* __restrict__ W1 = ty ? W1b : W1c;
    unsigned* __restrict__ dst =
        ws + (size_t)ty * W1_FRAG_DWORDS + ((size_t)((kt * 16 + nt) * 32 + lane)) * 8;

#pragma unroll
    for (int j = 0; j < 8; ++j) {
        int k0 = kt * 32 + 16 * half + 2 * j;
        dst[j] = packbf2(W1[(size_t)k0 * HID + n], W1[(size_t)(k0 + 1) * HID + n]);
    }
}

// ---------------------------------------------------------------------------
// Main kernel: blockDim = 128 (4 waves). blockIdx.x = 64-edge tile,
// blockIdx.y = edge type.
// ---------------------------------------------------------------------------
__global__ void __launch_bounds__(128)
edgemlp_kernel(const float* __restrict__ user_embed,
               const float* __restrict__ item_embed,
               const int* __restrict__ u_clicks, const int* __restrict__ v_clicks,
               const int* __restrict__ u_buys,   const int* __restrict__ v_buys,
               const float* __restrict__ b1_c, const float* __restrict__ W2_c,
               const float* __restrict__ b2_c,
               const float* __restrict__ b1_b, const float* __restrict__ W2_b,
               const float* __restrict__ b2_b,
               const uint4* __restrict__ wsB,   // bf16 W1 fragments (both types)
               float* __restrict__ out, int E) {
    // ldsA[(mt*8 + kt)*32 + lane] : one 32-byte A-fragment slice per lane
    __shared__ uint4 ldsA[M_TILES * 8 * 32 * 2];   // 32 KB
    __shared__ float ldsOut[4][M_TILES * 16];      // per-wave partials, 1 KB

    const int tid  = threadIdx.x;
    const int w    = tid >> 5;
    const int lane = tid & 31;
    const int half = lane >> 4;
    const int ncol = lane & 15;

    const int ty   = blockIdx.y;
    const int base = blockIdx.x * (M_TILES * 16);

    const int*   __restrict__ uidx = ty ? u_buys : u_clicks;
    const int*   __restrict__ vidx = ty ? v_buys : v_clicks;
    const float* __restrict__ b1p  = ty ? b1_b : b1_c;
    const float* __restrict__ w2p  = ty ? W2_b : W2_c;
    const float  b2v               = ty ? b2_b[0] : b2_c[0];
    const uint4* __restrict__ Bfrag = wsB + (size_t)ty * (W1_FRAG_DWORDS / 4);
    float* __restrict__ outp       = out + (size_t)ty * (size_t)E;

    // ---- gather + convert: 64 edges x 256 feats, 512 chunk-tasks of 32 f32 ----
#pragma unroll
    for (int it = 0; it < 4; ++it) {
        int task = tid + it * 128;
        int mg   = task >> 3;        // edge within tile, 0..63
        int c    = task & 7;         // k-tile (32 feats): 0..3 user, 4..7 item
        int e    = base + mg;
        if (e >= E) e = E - 1;       // clamped gather (stores masked later)

        const float* __restrict__ src =
            (c < 4) ? (user_embed + (size_t)uidx[e] * IN_DIM + c * 32)
                    : (item_embed + (size_t)vidx[e] * IN_DIM + (c - 4) * 32);

        float4 f[8];
#pragma unroll
        for (int j = 0; j < 8; ++j) f[j] = ((const float4*)src)[j];

        unsigned d[16];
#pragma unroll
        for (int q = 0; q < 8; ++q) {
            d[2 * q]     = packbf2(f[q].x, f[q].y);
            d[2 * q + 1] = packbf2(f[q].z, f[q].w);
        }

        // A-fragment scatter (contiguous k order, matching the W1 B-fragments):
        //   lane m    : i0..15 <- K 0..15
        //   lane m+16 : i0..15 <- K 16..31
        int mt = mg >> 4, m = mg & 15;
        uint4* A0 = &ldsA[((mt * 8 + c) * 32 + m) * 2];
        uint4* A1 = &ldsA[((mt * 8 + c) * 32 + m + 16) * 2];
        A0[0] = make_uint4(d[0], d[1], d[2], d[3]);
        A0[1] = make_uint4(d[4], d[5], d[6], d[7]);
        A1[0] = make_uint4(d[8], d[9], d[10], d[11]);
        A1[1] = make_uint4(d[12], d[13], d[14], d[15]);
    }

    // per-lane epilogue constants: column n = nt*16 + ncol, nt = w + 4*ni
    float b1v[NT_PER_WAVE], w2v[NT_PER_WAVE];
#pragma unroll
    for (int ni = 0; ni < NT_PER_WAVE; ++ni) {
        int n = (w + 4 * ni) * 16 + ncol;
        b1v[ni] = b1p[n];
        w2v[ni] = w2p[n];
    }

    __syncthreads();

    // ---- WMMA main loop: acc[nt][mt] fp32, bf16 A (LDS) x bf16 B (L2) ----
    v8f zero = {0.f, 0.f, 0.f, 0.f, 0.f, 0.f, 0.f, 0.f};
    v8f acc[NT_PER_WAVE][M_TILES];
#pragma unroll
    for (int ni = 0; ni < NT_PER_WAVE; ++ni)
#pragma unroll
        for (int mt = 0; mt < M_TILES; ++mt) acc[ni][mt] = zero;

#pragma unroll
    for (int kt = 0; kt < 8; ++kt) {
        ABFrag a[M_TILES];
#pragma unroll
        for (int mt = 0; mt < M_TILES; ++mt) {
            const uint4* ap = &ldsA[((mt * 8 + kt) * 32 + lane) * 2];
            a[mt].q[0] = ap[0];
            a[mt].q[1] = ap[1];
        }
#pragma unroll
        for (int ni = 0; ni < NT_PER_WAVE; ++ni) {
            int nt = w + 4 * ni;
            ABFrag b;
            const uint4* bp = Bfrag + (size_t)((kt * 16 + nt) * 32 + lane) * 2;
            b.q[0] = bp[0];
            b.q[1] = bp[1];
#pragma unroll
            for (int mt = 0; mt < M_TILES; ++mt) {
                acc[ni][mt] = __builtin_amdgcn_wmma_f32_16x16x32_bf16(
                    /*neg_a=*/false, a[mt].bf, /*neg_b=*/false, b.bf,
                    /*c_mod=*/(short)0, acc[ni][mt],
                    /*reuse_a=*/false, /*reuse_b=*/false);
            }
        }
    }

    // ---- fused layer 2: relu(acc + b1) . W2, reduce over lanes & waves ----
    // C/D layout: VGPR r -> row m = 8*half + r, col n = nt*16 + ncol.
#pragma unroll
    for (int mt = 0; mt < M_TILES; ++mt) {
#pragma unroll
        for (int r = 0; r < 8; ++r) {
            float s = 0.f;
#pragma unroll
            for (int ni = 0; ni < NT_PER_WAVE; ++ni) {
                float h = acc[ni][mt][r] + b1v[ni];
                h = h > 0.f ? h : 0.f;
                s += h * w2v[ni];
            }
            // reduce across the 16 lanes of this half-group
            s += __shfl_xor(s, 1, 32);
            s += __shfl_xor(s, 2, 32);
            s += __shfl_xor(s, 4, 32);
            s += __shfl_xor(s, 8, 32);
            if (ncol == 0) ldsOut[w][mt * 16 + half * 8 + r] = s;
        }
    }

    __syncthreads();

    if (tid < M_TILES * 16) {
        int e = base + tid;
        if (e < E) {
            outp[e] = ldsOut[0][tid] + ldsOut[1][tid] + ldsOut[2][tid] +
                      ldsOut[3][tid] + b2v;
        }
    }
}

extern "C" void kernel_launch(void* const* d_in, const int* in_sizes, int n_in,
                              void* d_out, int out_size, void* d_ws, size_t ws_size,
                              hipStream_t stream) {
    const float* user_embed = (const float*)d_in[0];
    const float* item_embed = (const float*)d_in[1];
    const int*   u_clicks   = (const int*)d_in[2];
    const int*   v_clicks   = (const int*)d_in[3];
    const int*   u_buys     = (const int*)d_in[4];
    const int*   v_buys     = (const int*)d_in[5];
    const float* W1_clicks  = (const float*)d_in[6];
    const float* b1_clicks  = (const float*)d_in[7];
    const float* W2_clicks  = (const float*)d_in[8];
    const float* b2_clicks  = (const float*)d_in[9];
    const float* W1_buys    = (const float*)d_in[10];
    const float* b1_buys    = (const float*)d_in[11];
    const float* W2_buys    = (const float*)d_in[12];
    const float* b2_buys    = (const float*)d_in[13];

    const int E = in_sizes[2];

    // 1) convert both W1 matrices to bf16 B-fragment layout in d_ws (256 KB)
    edgemlp_prep_w1<<<32, 256, 0, stream>>>(W1_clicks, W1_buys, (unsigned*)d_ws);

    // 2) main fused gather + GEMM + MLP kernel; y-dim = edge type
    int tiles = (E + 63) / 64;
    dim3 grid(tiles, 2);
    edgemlp_kernel<<<grid, 128, 0, stream>>>(
        user_embed, item_embed,
        u_clicks, v_clicks, u_buys, v_buys,
        b1_clicks, W2_clicks, b2_clicks,
        b1_buys, W2_buys, b2_buys,
        (const uint4*)d_ws, (float*)d_out, E);
}
